// S2ILayer2Model_IAS_47949014893103
// MI455X (gfx1250) — matrix operand order repaired and checked by reading, not compiled, not run on hardware
//
#include <hip/hip_runtime.h>
#include <hip/hip_bf16.h>
#include <math.h>

// ---- problem constants ----
#define B_   128
#define C_   200
#define A_   312
#define AP_  320          // A padded to 10*32
#define D_   2048
#define H_   1600
#define CPP_ 224          // C padded to 14*16
#define CT2_ 7            // pairs of 16-row tiles
#define TMPV 5.0f
#define LDSW 40           // LDS B-tile row stride (elements): 80B -> conflict-free b128

typedef __bf16 bf16_t;
typedef __attribute__((ext_vector_type(16))) __bf16 v16bf;
typedef __attribute__((ext_vector_type(8)))  __bf16 v8bf;
typedef __attribute__((ext_vector_type(8)))  float  v8f;

#define CAT16(lo, hi) __builtin_shufflevector(lo, hi, 0,1,2,3,4,5,6,7,8,9,10,11,12,13,14,15)

// ======================= small prep kernels =======================

__global__ void zero_f32(float* p, int n) {
    int i = blockIdx.x * 256 + threadIdx.x;
    if (i < n) p[i] = 0.0f;
}

__global__ void cvt_f32_bf16(const float* __restrict__ src, bf16_t* __restrict__ dst, int n) {
    int i = blockIdx.x * 256 + threadIdx.x;
    if (i < n) dst[i] = (bf16_t)src[i];
}

// W1 [H][A] -> bf16 [H][AP_] with zero-padded K tail
__global__ void cvt_w1_pad(const float* __restrict__ W1, bf16_t* __restrict__ W1b) {
    int i = blockIdx.x * 256 + threadIdx.x;         // over H_*AP_
    if (i >= H_ * AP_) return;
    int h = i / AP_, a = i - h * AP_;
    W1b[i] = (a < A_) ? (bf16_t)W1[(size_t)h * A_ + a] : (bf16_t)0.0f;
}

// comb[b][c][a] = att1[b,a] * AttM[c,a], zero-padded to [B][CPP_][AP_], bf16
__global__ void comb_kernel(const float* __restrict__ AttM, const float* __restrict__ att1,
                            bf16_t* __restrict__ comb) {
    long long i = (long long)blockIdx.x * 256 + threadIdx.x;   // over B_*CPP_*AP_
    if (i >= (long long)B_ * CPP_ * AP_) return;
    int a  = (int)(i % AP_);
    int bc = (int)(i / AP_);
    int c  = bc % CPP_;
    int b  = bc / CPP_;
    float v = 0.0f;
    if (a < A_ && c < C_) v = att1[b * A_ + a] * AttM[(size_t)c * A_ + a];
    comb[i] = (bf16_t)v;
}

// x = l2norm(x_in) per row of length D
__global__ void xnorm_kernel(const float* __restrict__ x_in, float* __restrict__ x) {
    int b = blockIdx.x, tid = threadIdx.x;
    __shared__ float red[256];
    float s = 0.f;
    for (int d = tid; d < D_; d += 256) { float v = x_in[b * D_ + d]; s += v * v; }
    red[tid] = s; __syncthreads();
    for (int st = 128; st > 0; st >>= 1) { if (tid < st) red[tid] += red[tid + st]; __syncthreads(); }
    float inv = 1.0f / fmaxf(sqrtf(red[0]), 1e-12f);
    for (int d = tid; d < D_; d += 256) x[b * D_ + d] = x_in[b * D_ + d] * inv;
}

// att_raw[b,a] = relu(x[b]·Wg[a] + bg[a])
__global__ void att_kernel(const float* __restrict__ x, const float* __restrict__ Wg,
                           const float* __restrict__ bg, float* __restrict__ att) {
    int b = blockIdx.x;
    int a = blockIdx.y * blockDim.x + threadIdx.x;
    if (a >= A_) return;
    const float* xr = x + (size_t)b * D_;
    const float* wr = Wg + (size_t)a * D_;
    float s = 0.f;
    for (int d = 0; d < D_; d++) s += xr[d] * wr[d];
    s += bg[a];
    att[b * A_ + a] = s > 0.f ? s : 0.f;
}

// att1 = softmax(att/TMP, axis=1) + 1   (in place)
__global__ void softmax_kernel(float* __restrict__ att) {
    int b = blockIdx.x, tid = threadIdx.x;   // blockDim = 512
    __shared__ float red[512];
    float v = (tid < A_) ? att[b * A_ + tid] * (1.0f / TMPV) : -3.0e38f;
    red[tid] = v; __syncthreads();
    for (int st = 256; st > 0; st >>= 1) { if (tid < st) red[tid] = fmaxf(red[tid], red[tid + st]); __syncthreads(); }
    float mx = red[0]; __syncthreads();
    float e = (tid < A_) ? __expf(v - mx) : 0.f;
    red[tid] = e; __syncthreads();
    for (int st = 256; st > 0; st >>= 1) { if (tid < st) red[tid] += red[tid + st]; __syncthreads(); }
    float sum = red[0];
    if (tid < A_) att[b * A_ + tid] = e / sum + 1.0f;
}

// ======================= WMMA fragment loaders =======================
// 16-bit A-matrix 16x32 (contiguous K, matching B): lanes 0-15: M=lane, K=0..15; lanes 16-31: M=lane-16, K=16..31
__device__ __forceinline__ v16bf load_afrag(const bf16_t* __restrict__ rowptr, int kbase, int lane) {
    int kb = (lane & 16) ? 16 : 0;
    v8bf lo = *(const v8bf*)(rowptr + kbase + kb);        // K = kb .. kb+7
    v8bf hi = *(const v8bf*)(rowptr + kbase + kb + 8);    // K = kb+8 .. kb+15
    return CAT16(lo, hi);
}

// 16-bit B-matrix 32x16 from global (contiguous K, matching A): lanes 0-15: N=lane, K=0..15; lanes 16-31: N=lane-16, K=16..31
__device__ __forceinline__ v16bf load_bfrag(const bf16_t* __restrict__ src, int ntile,
                                            int kbase, int ld, int lane) {
    int n  = lane & 15;
    int kb = (lane & 16) ? 16 : 0;
    const bf16_t* row = src + (size_t)(ntile * 16 + n) * ld + kbase + kb;
    v8bf lo = *(const v8bf*)(row);
    v8bf hi = *(const v8bf*)(row + 8);
    return CAT16(lo, hi);
}

// Same B fragment pattern, from an LDS tile laid out [64 rows][LDSW elems]
__device__ __forceinline__ v16bf lds_bfrag(const bf16_t* lb, int jtile, int lane) {
    int n  = lane & 15;
    int kb = (lane & 16) ? 16 : 0;
    const bf16_t* row = lb + (jtile * 16 + n) * LDSW + kb;
    v8bf lo = *(const v8bf*)(row);
    v8bf hi = *(const v8bf*)(row + 8);
    return CAT16(lo, hi);
}

#define WMMA_BF16(A, Bf, Cc) \
    __builtin_amdgcn_wmma_f32_16x16x32_bf16(false, (A), false, (Bf), (short)0, (Cc), false, false)

// ======================= GEMM 1: h = relu(comb @ W1^T + b1) =======================
// grid (B, CT2_, 25), block 32; each wave: 32 rows x 64 cols, K = AP_ = 320 (10 steps)
__global__ __launch_bounds__(32)
void h_gemm_kernel(const bf16_t* __restrict__ comb, const bf16_t* __restrict__ W1b,
                   const float* __restrict__ b1, bf16_t* __restrict__ hbuf) {
    int b = blockIdx.x, ct = blockIdx.y, ng = blockIdx.z;
    int lane = threadIdx.x;

    v8f acc[2][4];
#pragma unroll
    for (int t = 0; t < 2; t++)
#pragma unroll
        for (int j = 0; j < 4; j++)
#pragma unroll
            for (int r = 0; r < 8; r++) acc[t][j][r] = 0.f;

    int m = lane & 15;
    const bf16_t* arow0 = comb + (size_t)b * CPP_ * AP_ + (size_t)(ct * 32 + m) * AP_;
    const bf16_t* arow1 = arow0 + (size_t)16 * AP_;

    v16bf af0 = load_afrag(arow0, 0, lane);
    v16bf af1 = load_afrag(arow1, 0, lane);
    v16bf bf[4];
#pragma unroll
    for (int j = 0; j < 4; j++) bf[j] = load_bfrag(W1b, ng * 4 + j, 0, AP_, lane);

#pragma unroll
    for (int ks = 0; ks < 9; ks++) {            // pipelined: prefetch ks+1 before WMMAs of ks
        int kn = (ks + 1) * 32;
        v16bf af0n = load_afrag(arow0, kn, lane);
        v16bf af1n = load_afrag(arow1, kn, lane);
        v16bf bfn[4];
#pragma unroll
        for (int j = 0; j < 4; j++) bfn[j] = load_bfrag(W1b, ng * 4 + j, kn, AP_, lane);
#pragma unroll
        for (int j = 0; j < 4; j++) {
            acc[0][j] = WMMA_BF16(af0, bf[j], acc[0][j]);
            acc[1][j] = WMMA_BF16(af1, bf[j], acc[1][j]);
        }
        af0 = af0n; af1 = af1n;
#pragma unroll
        for (int j = 0; j < 4; j++) bf[j] = bfn[j];
    }
#pragma unroll
    for (int j = 0; j < 4; j++) {               // final k-step
        acc[0][j] = WMMA_BF16(af0, bf[j], acc[0][j]);
        acc[1][j] = WMMA_BF16(af1, bf[j], acc[1][j]);
    }

    int m0 = (lane & 16) ? 8 : 0;
    int n  = lane & 15;
#pragma unroll
    for (int t = 0; t < 2; t++)
#pragma unroll
        for (int j = 0; j < 4; j++) {
            int hcol = (ng * 4 + j) * 16 + n;
            float bias = b1[hcol];
#pragma unroll
            for (int r = 0; r < 8; r++) {
                int cr = ct * 32 + t * 16 + r + m0;     // < 224 always
                float v = acc[t][j][r] + bias;
                v = v > 0.f ? v : 0.f;
                hbuf[(size_t)b * CPP_ * H_ + (size_t)cr * H_ + hcol] = (bf16_t)v;
            }
        }
}

// ============ GEMM 2 (fused): cls = relu(h @ W2^T + b2); accumulate Σcls², Σcls·x ============
// grid (B, 1, 32), block 224 (7 waves). Block covers rows 0..223 x 64 cols; B tile shared via LDS.
// Per k-step: 128 threads stage the 64x32 bf16 B tile into a double-buffered LDS tile.
__global__ __launch_bounds__(224)
void cls_gemm_kernel(const bf16_t* __restrict__ hbuf, const bf16_t* __restrict__ W2b,
                     const float* __restrict__ b2, const float* __restrict__ x,
                     float* __restrict__ normsq, float* __restrict__ dotv) {
    int b = blockIdx.x, ng = blockIdx.z;
    int tid  = threadIdx.x;
    int lane = tid & 31;
    int wv   = tid >> 5;                 // 0..6 : M-pair index

    __shared__ __align__(16) bf16_t lbs[2][64 * LDSW];

    // --- staging assignment (threads 0..127): half-row each ---
    int srow  = tid >> 1;                // 0..63  (d-local)
    int shalf = (tid & 1) * 16;          // 0 or 16
    const bf16_t* gsrc = W2b + (size_t)(ng * 64 + srow) * H_ + shalf;
    bf16_t* lrow0 = &lbs[0][srow * LDSW + shalf];
    bf16_t* lrow1 = &lbs[1][srow * LDSW + shalf];

    v8f acc[2][4];
#pragma unroll
    for (int t = 0; t < 2; t++)
#pragma unroll
        for (int j = 0; j < 4; j++)
#pragma unroll
            for (int r = 0; r < 8; r++) acc[t][j][r] = 0.f;

    int m = lane & 15;
    const bf16_t* arow0 = hbuf + (size_t)b * CPP_ * H_ + (size_t)(wv * 32 + m) * H_;
    const bf16_t* arow1 = arow0 + (size_t)16 * H_;

    // stage k-step 0 into buffer 0
    if (tid < 128) {
        v8bf t0 = *(const v8bf*)(gsrc);
        v8bf t1 = *(const v8bf*)(gsrc + 8);
        *(v8bf*)(lrow0)     = t0;
        *(v8bf*)(lrow0 + 8) = t1;
    }
    // A fragments for k-step 0 (register-pipelined from global)
    v16bf af0 = load_afrag(arow0, 0, lane);
    v16bf af1 = load_afrag(arow1, 0, lane);
    __syncthreads();

    for (int ks = 0; ks < 50; ks++) {
        int p = ks & 1;
        // prefetch next B half-row into registers, next A fragments into registers
        v8bf t0, t1;
        v16bf af0n, af1n;
        bool more = (ks < 49);
        if (more) {
            if (tid < 128) {
                const bf16_t* g = gsrc + (ks + 1) * 32;
                t0 = *(const v8bf*)(g);
                t1 = *(const v8bf*)(g + 8);
            }
            af0n = load_afrag(arow0, (ks + 1) * 32, lane);
            af1n = load_afrag(arow1, (ks + 1) * 32, lane);
        }
        // compute from LDS buffer p
#pragma unroll
        for (int j = 0; j < 4; j++) {
            v16bf bfr = lds_bfrag(&lbs[p][0], j, lane);
            acc[0][j] = WMMA_BF16(af0, bfr, acc[0][j]);
            acc[1][j] = WMMA_BF16(af1, bfr, acc[1][j]);
        }
        if (more) {
            if (tid < 128) {
                bf16_t* l = p ? lrow0 : lrow1;     // write the other buffer
                *(v8bf*)(l)     = t0;
                *(v8bf*)(l + 8) = t1;
            }
            af0 = af0n; af1 = af1n;
        }
        __syncthreads();
    }

    // fused epilogue: relu + per-row partial Σcls² and Σcls·x over this wave's 64 columns
    int n  = lane & 15;
    int m0 = (lane & 16) ? 8 : 0;
#pragma unroll
    for (int t = 0; t < 2; t++) {
        float psn[8], psd[8];
#pragma unroll
        for (int r = 0; r < 8; r++) { psn[r] = 0.f; psd[r] = 0.f; }
#pragma unroll
        for (int j = 0; j < 4; j++) {
            int d = (ng * 4 + j) * 16 + n;
            float bias = b2[d];
            float xv   = x[(size_t)b * D_ + d];
#pragma unroll
            for (int r = 0; r < 8; r++) {
                float v = acc[t][j][r] + bias;
                v = v > 0.f ? v : 0.f;
                psn[r] += v * v;
                psd[r] += v * xv;
            }
        }
        // butterfly reduce across the 16 lanes of each half-wave
#pragma unroll
        for (int r = 0; r < 8; r++) {
            float a = psn[r], dd = psd[r];
            for (int off = 1; off < 16; off <<= 1) {
                a  += __shfl_xor(a,  off, 32);
                dd += __shfl_xor(dd, off, 32);
            }
            if ((lane & 15) == 0) {
                int c = wv * 32 + t * 16 + r + m0;
                if (c < C_) {
                    atomicAdd(&normsq[b * C_ + c], a);
                    atomicAdd(&dotv[b * C_ + c], dd);
                }
            }
        }
    }
}

// ======================= finalize =======================
__global__ void finalize_kernel(const float* __restrict__ dotv, const float* __restrict__ normsq,
                                const float* __restrict__ scale_cls, const float* __restrict__ bias_p,
                                float* __restrict__ out, int n) {
    int i = blockIdx.x * 256 + threadIdx.x;
    if (i >= n) return;
    float nv = fmaxf(sqrtf(normsq[i]), 1e-12f);
    out[i] = scale_cls[0] * (dotv[i] / nv + bias_p[0]);
}

// ======================= host launcher =======================
extern "C" void kernel_launch(void* const* d_in, const int* in_sizes, int n_in,
                              void* d_out, int out_size, void* d_ws, size_t ws_size,
                              hipStream_t stream) {
    (void)in_sizes; (void)n_in; (void)out_size; (void)ws_size;
    const float* AttM      = (const float*)d_in[0];
    const float* x_in      = (const float*)d_in[1];
    const float* Wg        = (const float*)d_in[2];
    const float* bg        = (const float*)d_in[3];
    const float* W1        = (const float*)d_in[4];
    const float* b1        = (const float*)d_in[5];
    const float* W2        = (const float*)d_in[6];
    const float* b2        = (const float*)d_in[7];
    const float* bias_p    = (const float*)d_in[8];
    const float* scale_cls = (const float*)d_in[9];
    float* out = (float*)d_out;

    char* ws = (char*)d_ws;
    auto alloc = [&](size_t bytes) -> char* {
        char* p = ws;
        ws += (bytes + 255) & ~(size_t)255;
        return p;
    };
    float*  x      = (float*) alloc((size_t)B_ * D_ * sizeof(float));
    float*  att    = (float*) alloc((size_t)B_ * A_ * sizeof(float));
    bf16_t* W1b    = (bf16_t*)alloc((size_t)H_ * AP_ * sizeof(bf16_t));
    bf16_t* W2b    = (bf16_t*)alloc((size_t)D_ * H_ * sizeof(bf16_t));
    bf16_t* comb   = (bf16_t*)alloc((size_t)B_ * CPP_ * AP_ * sizeof(bf16_t));
    bf16_t* hbuf   = (bf16_t*)alloc((size_t)B_ * CPP_ * H_ * sizeof(bf16_t));
    float*  normsq = (float*) alloc((size_t)B_ * C_ * sizeof(float));
    float*  dotv   = (float*) alloc((size_t)B_ * C_ * sizeof(float));

    // zero the atomic accumulators (every call — harness does not re-poison)
    zero_f32<<<(B_ * C_ + 255) / 256, 256, 0, stream>>>(normsq, B_ * C_);
    zero_f32<<<(B_ * C_ + 255) / 256, 256, 0, stream>>>(dotv,   B_ * C_);

    // x = l2norm(x_in)
    xnorm_kernel<<<B_, 256, 0, stream>>>(x_in, x);

    // att = relu(x @ Wg^T + bg); then softmax(att/TMP)+1 in place
    att_kernel<<<dim3(B_, (A_ + 255) / 256), 256, 0, stream>>>(x, Wg, bg, att);
    softmax_kernel<<<B_, 512, 0, stream>>>(att);

    // weights -> bf16 (W1 K-padded to 320)
    {
        int n1 = H_ * AP_;
        cvt_w1_pad<<<(n1 + 255) / 256, 256, 0, stream>>>(W1, W1b);
        int n2 = D_ * H_;
        cvt_f32_bf16<<<(n2 + 255) / 256, 256, 0, stream>>>(W2, W2b, n2);
    }

    // comb = (att+1) * AttM  -> bf16 [B][224][320], zero padded
    {
        long long n = (long long)B_ * CPP_ * AP_;
        comb_kernel<<<(int)((n + 255) / 256), 256, 0, stream>>>(AttM, att, comb);
    }

    // h = relu(comb @ W1^T + b1)  -> bf16 [B][224][1600]
    h_gemm_kernel<<<dim3(B_, CT2_, 25), 32, 0, stream>>>(comb, W1b, b1, hbuf);

    // cls GEMM (LDS-shared B tile) fused with Σcls² and Σcls·x reductions
    cls_gemm_kernel<<<dim3(B_, 1, 32), 224, 0, stream>>>(hbuf, W2b, b2, x, normsq, dotv);

    // out = scale * (dot / max(||cls||, eps) + bias)
    finalize_kernel<<<(B_ * C_ + 255) / 256, 256, 0, stream>>>(
        dotv, normsq, scale_cls, bias_p, out, B_ * C_);
}
